// GateAttention_66571993088393
// MI455X (gfx1250) — matrix operand order repaired and checked by reading, not compiled, not run on hardware
//
#include <hip/hip_runtime.h>
#include <hip/hip_bf16.h>

// ---------------------------------------------------------------------------
// GateAttention on gfx1250 (MI455X): bf16 WMMA pipeline, async-DMA pipelined
//   B=16, LC=2048, LQ=256, D=768, F=768
// ---------------------------------------------------------------------------

typedef __attribute__((ext_vector_type(16))) __bf16 v16bf;
typedef __attribute__((ext_vector_type(8)))  __bf16 v8bf;
typedef __attribute__((ext_vector_type(8)))  float  v8f;

#define B_  16
#define LC_ 2048
#define LQ_ 256
#define D_  768
#define F_  768

#define BM 128
#define BN 128
#define BK 32

// ---- gfx1250 async global->LDS path (GLOBAL_LOAD_ASYNC_TO_LDS_B128) --------
#if __has_builtin(__builtin_amdgcn_global_load_async_to_lds_b128) && \
    __has_builtin(__builtin_amdgcn_s_wait_asynccnt)
#define USE_ASYNC_LDS 1
#else
#define USE_ASYNC_LDS 0
#endif

typedef int v4i __attribute__((vector_size(16)));
typedef __attribute__((address_space(1))) v4i gv4i_t;   // global int4
typedef __attribute__((address_space(3))) v4i lv4i_t;   // LDS int4

// copy 16B global -> LDS (async DMA on gfx1250 when builtin available)
static __device__ __forceinline__ void copy16_g2l(const __bf16* g, __bf16* l) {
#if USE_ASYNC_LDS
  __builtin_amdgcn_global_load_async_to_lds_b128((gv4i_t*)g, (lv4i_t*)l, 0, 0);
#else
  *(v8bf*)l = *(const v8bf*)g;
#endif
}

static __device__ __forceinline__ void async_wait_all() {
#if USE_ASYNC_LDS
  __builtin_amdgcn_s_wait_asynccnt(0);
#endif
}

static __device__ __forceinline__ v16bf join16(v8bf lo, v8bf hi) {
  v16bf r;
#pragma unroll
  for (int i = 0; i < 8; ++i) { r[i] = lo[i]; r[i + 8] = hi[i]; }
  return r;
}

// ---------------------------------------------------------------------------
// f32 -> bf16 cast (grid-stride)
// ---------------------------------------------------------------------------
__global__ void cast_f32_to_bf16(const float* __restrict__ in,
                                 __bf16* __restrict__ out, long n) {
  long i = (long)blockIdx.x * blockDim.x + threadIdx.x;
  long stride = (long)gridDim.x * blockDim.x;
  for (; i < n; i += stride) out[i] = (__bf16)in[i];
}

// ---------------------------------------------------------------------------
// Generic bf16 WMMA GEMM: out = act( A @ Bm ), A row-major [M,K], Bm [K,N].
// Split-K concat: rows k < K0 come from A0 (ld=K0), k >= K0 from A1 (ld=K-K0).
// act: 0 = none (bf16 out), 1 = relu (bf16 out), 2 = sigmoid (f32 out).
// Batched via blockIdx.z and element strides bsA/bsB/bsO.
// Block: 256 threads = 8 waves as 4(M) x 2(N); block tile 128x128; K-step 32.
// Double-buffered LDS: tile t+1's async DMA overlaps tile t's WMMA work.
// ---------------------------------------------------------------------------
__global__ void __launch_bounds__(256)
gemm_bf16_wmma(const __bf16* __restrict__ A0,
               const __bf16* __restrict__ A1, int K0,
               const __bf16* __restrict__ Bm,
               void* __restrict__ outp,
               int M, int N, int K,
               long bsA, long bsB, long bsO,
               int act)
{
  __shared__ __bf16 As[2][BM][BK];         // 2 x 8 KB, [m][k] k-contiguous
  __shared__ __bf16 Bts[2][BN][BK + 8];    // 2 x 10 KB, [n][k] (+pad)

  const int bz = blockIdx.z;
  const __bf16* a0 = A0 ? A0 + bz * bsA : nullptr;
  const __bf16* a1 = A1 ? A1 + bz * bsA : nullptr;
  const __bf16* bmat = Bm + bz * bsB;

  const int bm   = blockIdx.x * BM;
  const int bn   = blockIdx.y * BN;
  const int tid  = threadIdx.x;
  const int lane = tid & 31;
  const int wave = tid >> 5;
  const int wm   = (wave & 3) * 32;        // wave tile: 32(M) x 64(N)
  const int wn   = (wave >> 2) * 64;

  // ---- loop-invariant staging pointers -------------------------------------
  // A tile: 512 x 16B chunks; thread handles chunks tid and tid+256
  const int srow = tid >> 2;               // 0..63
  const int sc8  = (tid & 3) * 8;          // 0,8,16,24
  const int la1  = K - K0;
  const __bf16* pA0 = a0 ? a0 + (long)(bm + srow) * K0  + sc8 : nullptr;
  const __bf16* pA1 = a1 ? a1 + (long)(bm + srow) * la1 + sc8 : nullptr;
  // B tile: thread handles chunks (kr=tid>>4, n8) and (kr+16, n8)
  const int skr = tid >> 4;                // 0..15
  const int sn8 = (tid & 15) * 8;          // 0..120
  const __bf16* pB = bmat + (long)skr * N + bn + sn8;

  // issue async DMA of tile t's A slice into buffer buf
  auto stageA = [&](int k0, int buf) {
    const bool first = (k0 < K0);
    const __bf16* aP = first ? (pA0 + k0) : (pA1 + (k0 - K0));
    const long aStp  = (long)64 * (first ? K0 : la1);
    copy16_g2l(aP,        &As[buf][srow][sc8]);
    copy16_g2l(aP + aStp, &As[buf][srow + 64][sc8]);
  };
  auto loadB = [&](int k0, v8bf& r0, v8bf& r1) {
    const __bf16* bP = pB + (long)k0 * N;
    r0 = *(const v8bf*)bP;
    r1 = *(const v8bf*)(bP + 16 * N);
  };

  v8f acc[2][4] = {};
  v8bf b0v, b1v;

  // prologue: tile 0 in flight
  loadB(0, b0v, b1v);
  stageA(0, 0);

  const int T = K / BK;
  for (int t = 0; t < T; ++t) {
    const int cur = t & 1;

    // finish tile t staging: transpose-store B regs into LDS
#pragma unroll
    for (int i = 0; i < 8; ++i) {
      Bts[cur][sn8 + i][skr]      = b0v[i];
      Bts[cur][sn8 + i][skr + 16] = b1v[i];
    }
    async_wait_all();                      // A(t) DMA complete
    __syncthreads();                       // tile t visible; buf cur^1 free

    // kick off tile t+1 into the other buffer (overlaps WMMA below)
    if (t + 1 < T) {
      loadB((t + 1) * BK, b0v, b1v);       // held in regs through compute
      stageA((t + 1) * BK, cur ^ 1);       // async DMA, no wait here
      if ((t + 2) * BK < K)                // speculative prefetch of t+2 B
        __builtin_prefetch(pB + (long)(t + 2) * BK * N, 0, 0);
    }

    // ---- WMMA on tile t: fragment layouts per CDNA5 ISA 7.12.2 ----
#pragma unroll
    for (int mt = 0; mt < 2; ++mt) {
      // A 16x32 bf16: lanes 0-15 row=lane, K 0..7 & 16..23; lanes 16-31 K 8..15 & 24..31
      int mr = wm + mt * 16 + (lane & 15);
      int ks = (lane >> 4) * 8;
      v16bf afrag = join16(*(v8bf*)&As[cur][mr][ks], *(v8bf*)&As[cur][mr][ks + 16]);
#pragma unroll
      for (int nt = 0; nt < 4; ++nt) {
        // B 32x16 bf16: lanes 0-15 col=lane, K 0..7 & 16..23; lanes 16-31 K 8..15 & 24..31
        int nr = wn + nt * 16 + (lane & 15);
        int kf = (lane >> 4) * 8;
        v16bf bfrag = join16(*(v8bf*)&Bts[cur][nr][kf], *(v8bf*)&Bts[cur][nr][kf + 16]);
        acc[mt][nt] = __builtin_amdgcn_wmma_f32_16x16x32_bf16(
            false, afrag, false, bfrag, (short)0, acc[mt][nt], false, false);
      }
    }
    // no trailing barrier needed: next iteration writes only buf cur^1,
    // whose readers all passed this iteration's barrier already.
  }

  // ---- epilogue: C/D layout — VGPR r, lanes 0-15 M=r, lanes 16-31 M=r+8 ----
  const int mhi = (lane >> 4) << 3;
  if (act == 2) {
    float* outF = (float*)outp + bz * bsO;
#pragma unroll
    for (int mt = 0; mt < 2; ++mt)
#pragma unroll
      for (int nt = 0; nt < 4; ++nt) {
        int col = bn + wn + nt * 16 + (lane & 15);
#pragma unroll
        for (int r = 0; r < 8; ++r) {
          int row = bm + wm + mt * 16 + r + mhi;
          float v = acc[mt][nt][r];
          outF[(long)row * N + col] = 1.f / (1.f + __expf(-v));
        }
      }
  } else {
    __bf16* outB = (__bf16*)outp + bz * bsO;
#pragma unroll
    for (int mt = 0; mt < 2; ++mt)
#pragma unroll
      for (int nt = 0; nt < 4; ++nt) {
        int col = bn + wn + nt * 16 + (lane & 15);
#pragma unroll
        for (int r = 0; r < 8; ++r) {
          int row = bm + wm + mt * 16 + r + mhi;
          float v = acc[mt][nt][r];
          if (act == 1 && v < 0.f) v = 0.f;
          outB[(long)row * N + col] = (__bf16)v;
        }
      }
  }
}

// ---------------------------------------------------------------------------
// Fused logits + mask + softmax:
//   logits[b, m, q] = (x_cont_[b,m,:] . x_ques_[b,q,:]) / sqrt(F)  (+mask)
//   attn = softmax over q, stored bf16.
// Block = 64 threads (2 waves), covers 32 LC rows; each wave one 16-row tile
// against all 256 question positions (16 N-tiles). x_ques_ rows stage in LDS;
// since logits uses x_ques_^T, its rows ARE the WMMA B columns (no transpose).
// ---------------------------------------------------------------------------
#define AROWS 32

__global__ void __launch_bounds__(64)
attn_softmax(const __bf16* __restrict__ xcp,   // [B, LC, F]
             const __bf16* __restrict__ xqp,   // [B, LQ, F]
             const int* __restrict__ qlenp,    // [B]
             __bf16* __restrict__ attn)        // [B, LC, LQ]
{
  __shared__ __bf16 Bts[LQ_][BK + 8];          // 20 KB: xq' rows [n][k]
  __shared__ float  logits[AROWS][LQ_];        // 32 KB

  const int b       = blockIdx.y;
  const int rowBase = blockIdx.x * AROWS;
  const int tid     = threadIdx.x;
  const int lane    = tid & 31;
  const int wave    = tid >> 5;                // 0..1
  const long xqbase = (long)b * LQ_ * F_;
  const long xcbase = ((long)b * LC_ + rowBase) * F_;

  // loop-invariant staging pointers: thread covers rows (tid>>2) + 16*it
  const __bf16* gq0 = xqp + xqbase + (long)(tid >> 2) * F_ + (tid & 3) * 8;
  __bf16* lq0 = &Bts[tid >> 2][(tid & 3) * 8];

  v8f acc[16] = {};

  for (int k0 = 0; k0 < F_; k0 += BK) {
    // stage 256 x 32 slice of x_ques_ : 16 x 16B async chunks per thread
#pragma unroll
    for (int it = 0; it < 16; ++it)
      copy16_g2l(gq0 + k0 + (long)it * 16 * F_, lq0 + it * 16 * (BK + 8));
    if (k0 + BK < F_) {
      __builtin_prefetch(xcp + xcbase + (long)(wave * 16 + (lane & 15)) * F_ + k0 + BK, 0, 0);
    }
    async_wait_all();
    __syncthreads();

    // A fragment straight from global (rows are contiguous in K)
    int mr = wave * 16 + (lane & 15);
    int ks = (lane >> 4) * 8;
    const __bf16* ap = xcp + xcbase + (long)mr * F_ + k0;
    v16bf afrag = join16(*(const v8bf*)(ap + ks), *(const v8bf*)(ap + ks + 16));
#pragma unroll
    for (int nt = 0; nt < 16; ++nt) {
      int nr = nt * 16 + (lane & 15);
      int kf = (lane >> 4) * 8;
      v16bf bfrag = join16(*(v8bf*)&Bts[nr][kf], *(v8bf*)&Bts[nr][kf + 16]);
      acc[nt] = __builtin_amdgcn_wmma_f32_16x16x32_bf16(
          false, afrag, false, bfrag, (short)0, acc[nt], false, false);
    }
    __syncthreads();
  }

  // dump accumulators to LDS (C/D layout), scaled by 1/sqrt(F)
  const float scale = 0.036084391824351615f;   // 1/sqrt(768)
  const int mhi = (lane >> 4) << 3;
#pragma unroll
  for (int nt = 0; nt < 16; ++nt)
#pragma unroll
    for (int r = 0; r < 8; ++r) {
      int lr = wave * 16 + r + mhi;
      int cq = nt * 16 + (lane & 15);
      logits[lr][cq] = acc[nt][r] * scale;
    }
  __syncthreads();

  // per-row masked softmax (one thread per LC row)
  const int qlen = qlenp[b];
  if (tid < AROWS) {
    float mx = -3.4e38f;
    for (int j = 0; j < LQ_; ++j) {
      float v = logits[tid][j];
      if (j >= qlen) v += -1.0e12f;            // MASK_VALUE
      logits[tid][j] = v;
      mx = fmaxf(mx, v);
    }
    float s = 0.f;
    for (int j = 0; j < LQ_; ++j) {
      float e = __expf(logits[tid][j] - mx);
      logits[tid][j] = e;
      s += e;
    }
    float inv = 1.f / s;
    __bf16* ao = attn + ((long)b * LC_ + rowBase + tid) * LQ_;
    for (int j = 0; j < LQ_; ++j) ao[j] = (__bf16)(logits[tid][j] * inv);
  }
}

// ---------------------------------------------------------------------------
// Host-side orchestration
// ---------------------------------------------------------------------------
extern "C" void kernel_launch(void* const* d_in, const int* in_sizes, int n_in,
                              void* d_out, int out_size, void* d_ws, size_t ws_size,
                              hipStream_t stream) {
  const float* x_cont = (const float*)d_in[0];   // [B, LC, D]
  const float* x_ques = (const float*)d_in[1];   // [B, LQ, D]
  const int*   qlen   = (const int*)d_in[2];     // [B, 1]
  const float* WC     = (const float*)d_in[3];   // [D, F]
  const float* WQ     = (const float*)d_in[4];   // [D, F]
  const float* V      = (const float*)d_in[5];   // [2D, F]

  // workspace layout (bytes)
  char* ws = (char*)d_ws;
  __bf16* WCb   = (__bf16*)(ws + 0);             //  1,179,648
  __bf16* WQb   = (__bf16*)(ws + 1179648);       //  1,179,648
  __bf16* Vb    = (__bf16*)(ws + 2359296);       //  2,359,296
  __bf16* xqb   = (__bf16*)(ws + 4718592);       //  6,291,456
  __bf16* xqpb  = (__bf16*)(ws + 11010048);      //  6,291,456
  __bf16* xcb   = (__bf16*)(ws + 17301504);      // 50,331,648
  __bf16* xcpb  = (__bf16*)(ws + 67633152);      // 50,331,648
  __bf16* attnb = (__bf16*)(ws + 117964800);     // 16,777,216
  __bf16* Cb    = (__bf16*)(ws + 134742016);     // 50,331,648  (end 185,073,664)

  auto cast = [&](const float* in, __bf16* out, long n) {
    long blocks = (n + 255) / 256;
    if (blocks > 4096) blocks = 4096;
    cast_f32_to_bf16<<<(int)blocks, 256, 0, stream>>>(in, out, n);
  };

  cast(x_cont, xcb, (long)B_ * LC_ * D_);
  cast(x_ques, xqb, (long)B_ * LQ_ * D_);
  cast(WC, WCb, (long)D_ * F_);
  cast(WQ, WQb, (long)D_ * F_);
  cast(V,  Vb,  (long)2 * D_ * F_);

  // x_ques_ = relu(x_ques @ WQ)  : M=4096, N=768, K=768
  gemm_bf16_wmma<<<dim3((B_ * LQ_) / BM, F_ / BN, 1), 256, 0, stream>>>(
      xqb, nullptr, D_, WQb, (void*)xqpb,
      B_ * LQ_, F_, D_, 0, 0, 0, /*act=*/1);

  // x_cont_ = relu(x_cont @ WC)  : M=32768, N=768, K=768
  gemm_bf16_wmma<<<dim3((B_ * LC_) / BM, F_ / BN, 1), 256, 0, stream>>>(
      xcb, nullptr, D_, WCb, (void*)xcpb,
      B_ * LC_, F_, D_, 0, 0, 0, /*act=*/1);

  // attn = softmax(mask(x_cont_ @ x_ques_^T / sqrt(F)))
  attn_softmax<<<dim3(LC_ / AROWS, B_, 1), 64, 0, stream>>>(
      xcpb, xqpb, qlen, attnb);

  // C = attn @ x_ques  (batched): M=2048, N=768, K=256 per batch
  gemm_bf16_wmma<<<dim3(LC_ / BM, D_ / BN, B_), 256, 0, stream>>>(
      attnb, nullptr, LQ_, xqb, (void*)Cb,
      LC_, D_, LQ_,
      (long)LC_ * LQ_, (long)LQ_ * D_, (long)LC_ * D_, /*act=*/0);

  // gate = sigmoid([x_cont, C] @ V) : M=32768, N=768, K=1536 (split at 768)
  gemm_bf16_wmma<<<dim3((B_ * LC_) / BM, F_ / BN, 1), 256, 0, stream>>>(
      xcb, Cb, D_, Vb, d_out,
      B_ * LC_, F_, 2 * D_, 0, 0, 0, /*act=*/2);
}
